// SpikeFP32Adder_23407571764146
// MI455X (gfx1250) — matrix operand order repaired and checked by reading, not compiled, not run on hardware
//
#include <hip/hip_runtime.h>

typedef __attribute__((ext_vector_type(16))) _Float16 v16h;
typedef __attribute__((ext_vector_type(8)))  float    v8f;

#define WAVES_PER_BLOCK 8

// Exact integer replication of the reference soft-logic FP32 adder circuit.
// Inputs/outputs are IEEE-754 bit patterns as uint32 (bit 31 = A[...,0]).
__device__ __forceinline__ unsigned fp32_add_circuit(unsigned a, unsigned b) {
  const unsigned s_a = a >> 31, s_b = b >> 31;
  const unsigned e_a = (a >> 23) & 0xFFu, e_b = (b >> 23) & 0xFFu;
  const unsigned mra = a & 0x7FFFFFu,     mrb = b & 0x7FFFFFu;
  const unsigned hid_a = (e_a != 0u), hid_b = (e_b != 0u);
  const unsigned ea = hid_a ? e_a : 1u, eb = hid_b ? e_b : 1u;
  const unsigned ma24 = (hid_a << 23) | mra, mb24 = (hid_b << 23) | mrb;

  const bool exp_gt = ea > eb, exp_eq = ea == eb;
  const bool mant_gt = ma24 > mb24, mant_eq = ma24 == mb24;
  const bool abs_eq = exp_eq && mant_eq;
  const bool a_ge_b = exp_gt || (exp_eq && (mant_gt || mant_eq));

  const unsigned exp_diff = a_ge_b ? (ea - eb) : (eb - ea);
  const bool big_diff = exp_diff >= 24u;          // high_any || (bit4 & bit3)
  const unsigned sh = exp_diff & 31u;             // exp_diff[3:8]

  const unsigned e_max     = a_ge_b ? ea : eb;
  const unsigned m_large   = (a_ge_b ? ma24 : mb24) << 4;   // 28-bit
  const unsigned m_small_u = (a_ge_b ? mb24 : ma24) << 4;
  const unsigned m_small   = m_small_u >> sh;
  const bool sticky_sh = (m_small_u & ((1u << sh) - 1u)) != 0u;

  const bool dsign  = (s_a ^ s_b) != 0u;
  const bool cancel = dsign && abs_eq;
  const unsigned s_large = a_ge_b ? s_a : s_b;

  const unsigned sum_full  = m_large + m_small;
  const unsigned sum_carry = (sum_full >> 28) & 1u;
  const unsigned sum28     = sum_full & 0x0FFFFFFFu;
  unsigned diff28 = (m_large - m_small) & 0x0FFFFFFFu;
  if (dsign && sticky_sh) diff28 = (diff28 - 1u) & 0x0FFFFFFFu;

  const unsigned mant = dsign ? diff28 : sum28;
  const unsigned rc   = dsign ? 0u : sum_carry;

  const unsigned lzc  = (mant == 0u) ? 28u : ((unsigned)__builtin_clz(mant) - 4u);
  const bool under    = lzc >= e_max;             // lzc_8 >= e_max
  const unsigned norm = (mant << lzc) & 0x0FFFFFFFu;
  const unsigned e_after = (e_max - lzc) & 0xFFu;
  const unsigned e_p1    = (e_max + 1u) & 0xFFu;
  const unsigned e_pre   = rc ? e_p1 : (under ? 0u : e_after);

  const unsigned m_ovf = mant >> 5;               // mant[0:23] MSB-first
  const unsigned r_ovf = (mant >> 4) & 1u;
  const bool     st_ovf = (mant & 0xFu) != 0u;
  const unsigned m_nrm = (norm >> 4) & 0x7FFFFFu; // norm[1:24]
  const unsigned r_nrm = (norm >> 3) & 1u;
  const bool     st_nrm = (norm & 7u) != 0u;

  const unsigned m_pre = rc ? m_ovf : m_nrm;
  const unsigned r_pre = rc ? r_ovf : r_nrm;
  const bool     st_pre = (rc ? st_ovf : st_nrm) || (!dsign && sticky_sh);
  const unsigned m_sel = under ? (mant >> 5) : m_pre;
  const unsigned Lb = m_sel & 1u;
  const unsigned do_round = (r_pre && (st_pre || Lb) && !under) ? 1u : 0u;
  // Reference's 24-bit rounding adder: carry-out is structurally 0, so the
  // exponent is never bumped by rounding overflow; mantissa wraps mod 2^23.
  const unsigned m_fin = (m_sel + do_round) & 0x7FFFFFu;
  const unsigned ce = e_pre;

  unsigned fs = cancel ? 0u : s_large;
  unsigned fe = cancel ? 0u : ce;
  unsigned fm = cancel ? 0u : m_fin;
  if (ce == 0xFFu) { fs = s_large; fe = 0xFFu; fm = 0u; }

  const unsigned normal = (fs << 31) | (fe << 23) | fm;
  const unsigned larger = a_ge_b ? a : b;
  return big_diff ? larger : normal;
}

// A-operand fragment for v_wmma_f32_16x16x32_f16 (wave32), in this file's contiguous K order:
// lane L holds row M=L&15, K in {h*16 .. h*16+15}, h=L>>4.
// Elements 0..7 = first chunk, 8..15 = second chunk (K pairs packed per VGPR).
__device__ __forceinline__ v16h load_frag(const float* __restrict__ src, int row, int lane) {
  const float* p = src + (size_t)row * 32u + (size_t)((lane >> 4) * 16);
  float4 c0 = *(const float4*)(p);
  float4 c1 = *(const float4*)(p + 4);
  float4 c2 = *(const float4*)(p + 8);
  float4 c3 = *(const float4*)(p + 12);
  v16h t;
  t[0]=(_Float16)c0.x;  t[1]=(_Float16)c0.y;  t[2]=(_Float16)c0.z;  t[3]=(_Float16)c0.w;
  t[4]=(_Float16)c1.x;  t[5]=(_Float16)c1.y;  t[6]=(_Float16)c1.z;  t[7]=(_Float16)c1.w;
  t[8]=(_Float16)c2.x;  t[9]=(_Float16)c2.y;  t[10]=(_Float16)c2.z; t[11]=(_Float16)c2.w;
  t[12]=(_Float16)c3.x; t[13]=(_Float16)c3.y; t[14]=(_Float16)c3.z; t[15]=(_Float16)c3.w;
  return t;
}

__global__ __launch_bounds__(WAVES_PER_BLOCK * 32)
void bitcircuit_fp32_add_kernel(const float* __restrict__ A,
                                const float* __restrict__ B,
                                float* __restrict__ out, int nrows) {
  // Per-wave scratch: [op(2)][tile(2)][col hi/lo(2)][row-half(2)][8] floats.
  __shared__ float lds[WAVES_PER_BLOCK * 128];
  const int lane = (int)(threadIdx.x & 31u);
  const int wv   = (int)(threadIdx.x >> 5u);
  const int lbase = wv * 128;
  const long long rowBase = ((long long)blockIdx.x * WAVES_PER_BLOCK + wv) * 32ll;
  const bool full = (rowBase + 32 <= (long long)nrows);   // wave-uniform

  const int m = lane & 15;
  const int t = lane >> 4;

  if (full) {
    // ---- constant weight fragment (B matrix 32x16 f16) ----
    // B layout used here (wave32, contiguous K matching load_frag): lanes 0-15 hold K=0..15 (col = lane),
    // lanes 16-31 hold K=16..31 (col = lane-16), K pairs packed per VGPR.
    // Col 0 = hi weights (K<16: 2^(15-K)); col 1 = lo weights (K>=16: 2^(31-K)).
    v16h wf;
#pragma unroll
    for (int e = 0; e < 16; ++e) wf[e] = (_Float16)0.0f;
    if (lane == 0 || lane == 17) {
#pragma unroll
      for (int e = 0; e < 16; ++e) wf[e] = (_Float16)(float)(1u << (15 - e));
    }
    const v8f zero = {0.f, 0.f, 0.f, 0.f, 0.f, 0.f, 0.f, 0.f};

    const int row0 = (int)rowBase + m;        // fragment rows this lane supplies
    const int row1 = (int)rowBase + 16 + m;

#pragma unroll
    for (int op = 0; op < 2; ++op) {
      const float* src = (op == 0) ? A : B;
      v16h f0 = load_frag(src, row0, lane);
      v16h f1 = load_frag(src, row1, lane);
      // Exact bit-packing: products are bit*2^p; every partial f32 sum is an
      // integer < 2^16, so accumulation is exact.
      v8f d0 = __builtin_amdgcn_wmma_f32_16x16x32_f16(false, f0, false, wf,
                                                      (short)0, zero, false, false);
      v8f d1 = __builtin_amdgcn_wmma_f32_16x16x32_f16(false, f1, false, wf,
                                                      (short)0, zero, false, false);
      // D layout: lanes {0,16} hold column N=0 (hi), lanes {1,17} hold N=1 (lo);
      // VGPR r = row M=r (+8 for lanes>=16). Spill those scalars to LDS.
      if (lane == 0 || lane == 1 || lane == 16 || lane == 17) {
        const int col = lane & 1, half = lane >> 4;
#pragma unroll
        for (int r = 0; r < 8; ++r) {
          lds[lbase + (((op * 2 + 0) * 2 + col) * 2 + half) * 8 + r] = d0[r];
          lds[lbase + (((op * 2 + 1) * 2 + col) * 2 + half) * 8 + r] = d1[r];
        }
      }
    }
  }

  // Every thread in the block reaches this barrier exactly once.
  __syncthreads();

  if (full) {
    const int ri = (m >> 3), ci = (m & 7);
    const float hi0 = lds[lbase + (((0 * 2 + t) * 2 + 0) * 2 + ri) * 8 + ci];
    const float lo0 = lds[lbase + (((0 * 2 + t) * 2 + 1) * 2 + ri) * 8 + ci];
    const float hi1 = lds[lbase + (((1 * 2 + t) * 2 + 0) * 2 + ri) * 8 + ci];
    const float lo1 = lds[lbase + (((1 * 2 + t) * 2 + 1) * 2 + ri) * 8 + ci];
    const unsigned packed0 = ((unsigned)hi0 << 16) | (unsigned)lo0;
    const unsigned packed1 = ((unsigned)hi1 << 16) | (unsigned)lo1;

    const unsigned res = fp32_add_circuit(packed0, packed1);
    float* orow = out + (size_t)(rowBase + lane) * 32u;
#pragma unroll
    for (int g = 0; g < 8; ++g) {
      float4 v;
      v.x = (float)((res >> (31 - 4 * g)) & 1u);
      v.y = (float)((res >> (30 - 4 * g)) & 1u);
      v.z = (float)((res >> (29 - 4 * g)) & 1u);
      v.w = (float)((res >> (28 - 4 * g)) & 1u);
      *(float4*)(orow + 4 * g) = v;
    }
  } else {
    // Tail path (no WMMA; per-lane scalar packing with bounds checks).
    const long long row = rowBase + lane;
    if (row < (long long)nrows) {
      const float* ar = A + (size_t)row * 32u;
      const float* br = B + (size_t)row * 32u;
      unsigned pa = 0u, pb = 0u;
#pragma unroll
      for (int i = 0; i < 32; ++i) {
        pa |= (ar[i] != 0.f ? 1u : 0u) << (31 - i);
        pb |= (br[i] != 0.f ? 1u : 0u) << (31 - i);
      }
      const unsigned res = fp32_add_circuit(pa, pb);
      float* orow = out + (size_t)row * 32u;
#pragma unroll
      for (int i = 0; i < 32; ++i) orow[i] = (float)((res >> (31 - i)) & 1u);
    }
  }
}

extern "C" void kernel_launch(void* const* d_in, const int* in_sizes, int n_in,
                              void* d_out, int out_size, void* d_ws, size_t ws_size,
                              hipStream_t stream) {
  (void)n_in; (void)out_size; (void)d_ws; (void)ws_size;
  const float* A = (const float*)d_in[0];
  const float* B = (const float*)d_in[1];
  float* out = (float*)d_out;
  const long long nrows  = (long long)in_sizes[0] / 32ll;   // 524288 for the reference
  const long long chunks = (nrows + 31) / 32;               // one 32-row chunk per wave
  const int blocks = (int)((chunks + WAVES_PER_BLOCK - 1) / WAVES_PER_BLOCK);
  if (blocks > 0) {
    bitcircuit_fp32_add_kernel<<<blocks, WAVES_PER_BLOCK * 32, 0, stream>>>(
        A, B, out, (int)nrows);
  }
}
